// RobertaSelfAttentionZIPTF_31318901523106
// MI455X (gfx1250) — 16-bit and 8-bit matrix operand order repaired and checked by reading, compiled, not run on hardware
//
#include <hip/hip_runtime.h>
#include <hip/hip_bf16.h>

typedef __attribute__((ext_vector_type(16))) _Float16 v16h;
typedef __attribute__((ext_vector_type(8)))  _Float16 v8h;
typedef __attribute__((ext_vector_type(8)))  float    v8f;
typedef __attribute__((ext_vector_type(8)))  int      v8i;
typedef int v4i_gcc __attribute__((vector_size(16)));

#define B_  4
#define S_  1024
#define D_  1024
#define H_  16
#define DH_ 64

#if defined(__has_builtin)
#  if __has_builtin(__builtin_amdgcn_global_load_async_to_lds_b128)
#    define HAVE_ASYNC_G2L 1
#  endif
#  if __has_builtin(__builtin_amdgcn_s_wait_asynccnt)
#    define HAVE_WAIT_ASYNC 1
#  endif
#endif

#ifdef HAVE_ASYNC_G2L
__device__ __forceinline__ void g2l_async_b128(const void* g, void* l) {
    __builtin_amdgcn_global_load_async_to_lds_b128(
        (v4i_gcc*)g, (v4i_gcc*)l, 0, 0);
}
__device__ __forceinline__ void wait_async0() {
#ifdef HAVE_WAIT_ASYNC
    __builtin_amdgcn_s_wait_asynccnt(0);
#else
    asm volatile("s_wait_asynccnt 0" ::: "memory");
#endif
}
#endif

__global__ __launch_bounds__(256) void cvt_f16_kernel(
    const float* __restrict__ in, _Float16* __restrict__ out, int n)
{
    const int i = (blockIdx.x * 256 + threadIdx.x) * 8;
    if (i + 8 <= n) {
        const float4 x = *(const float4*)(in + i);
        const float4 y = *(const float4*)(in + i + 4);
        v8h h;
        h[0] = (_Float16)x.x; h[1] = (_Float16)x.y;
        h[2] = (_Float16)x.z; h[3] = (_Float16)x.w;
        h[4] = (_Float16)y.x; h[5] = (_Float16)y.y;
        h[6] = (_Float16)y.z; h[7] = (_Float16)y.w;
        *(v8h*)(out + i) = h;
    }
}

__global__ __launch_bounds__(128) void qkv_gemm_kernel(
    const _Float16* __restrict__ Ah,
    const _Float16* __restrict__ WqH, const float* __restrict__ bq,
    const _Float16* __restrict__ WkH, const float* __restrict__ bk,
    const _Float16* __restrict__ WvH, const float* __restrict__ bv,
    signed char* __restrict__ qhat, signed char* __restrict__ khat,
    signed char* __restrict__ vhatT)
{
    __shared__ _Float16 lbs[2][64 * 32];

    const int tid  = threadIdx.x;
    const int wave = tid >> 5, lane = tid & 31;
    const int l16  = lane & 15, half = lane >> 4;
    const int hiA  = half * 8;
    const int hiB  = half * 8;

    const int ntile = blockIdx.x;
    const int mtile = blockIdx.y;
    const int which = blockIdx.z;
    const _Float16* W    = (which == 0) ? WqH : (which == 1) ? WkH : WvH;
    const float*    bias = (which == 0) ? bq  : (which == 1) ? bk  : bv;

    const int rowA = mtile * 64 + wave * 16 + l16;
    const _Float16* arow = Ah + (size_t)rowA * D_;

    const int n_loc = tid >> 1, seg = tid & 1;
    const _Float16* wsrc = W + (size_t)(ntile * 64 + n_loc) * D_ + seg * 16;
    _Float16* const ldst0 = &lbs[0][n_loc * 32 + seg * 16];
    _Float16* const ldst1 = &lbs[1][n_loc * 32 + seg * 16];

#define FILL_PANEL_ASYNC(ib, k0)                                   \
    do {                                                           \
        _Float16* d = (ib) ? ldst1 : ldst0;                        \
        g2l_async_b128(wsrc + (k0),     d);                        \
        g2l_async_b128(wsrc + (k0) + 8, d + 8);                    \
    } while (0)
#define FILL_PANEL_SYNC(ib, k0)                                    \
    do {                                                           \
        _Float16* d = (ib) ? ldst1 : ldst0;                        \
        v8h t0 = *(const v8h*)(wsrc + (k0));                       \
        v8h t1 = *(const v8h*)(wsrc + (k0) + 8);                   \
        *(v8h*)d = t0;                                             \
        *(v8h*)(d + 8) = t1;                                       \
    } while (0)

    v8f acc[4] = {};

#ifdef HAVE_ASYNC_G2L
    FILL_PANEL_ASYNC(0, 0);
    wait_async0();
#else
    FILL_PANEL_SYNC(0, 0);
#endif
    __syncthreads();

    int ib = 0;
    for (int k0 = 0; k0 < D_; k0 += 32, ib ^= 1) {
        if (k0 + 32 < D_) {
#ifdef HAVE_ASYNC_G2L
            FILL_PANEL_ASYNC(ib ^ 1, k0 + 32);
#else
            FILL_PANEL_SYNC(ib ^ 1, k0 + 32);
#endif
        }
        const v8h alo = *(const v8h*)(arow + k0 + hiA);
        const v8h ahi = *(const v8h*)(arow + k0 + hiA + 16);
        v16h a;
#pragma unroll
        for (int i = 0; i < 8; ++i) { a[i] = alo[i]; a[i + 8] = ahi[i]; }

        const _Float16* lb = ib ? lbs[1] : lbs[0];
#pragma unroll
        for (int t = 0; t < 4; ++t) {
            const _Float16* bp = lb + (t * 16 + l16) * 32 + hiB;
            const v8h b0 = *(const v8h*)bp;
            const v8h b1 = *(const v8h*)(bp + 16);
            v16h b;
#pragma unroll
            for (int i = 0; i < 8; ++i) { b[i] = b0[i]; b[i + 8] = b1[i]; }
            acc[t] = __builtin_amdgcn_wmma_f32_16x16x32_f16(
                         false, a, false, b, (short)0, acc[t], false, false);
        }
#ifdef HAVE_ASYNC_G2L
        wait_async0();
#endif
        __syncthreads();
    }

#pragma unroll
    for (int t = 0; t < 4; ++t) {
        const int gn = ntile * 64 + t * 16 + l16;
        const float bn = bias[gn];
#pragma unroll
        for (int j = 0; j < 8; ++j) {
            const int row = mtile * 64 + wave * 16 + j + half * 8;
            const float x = acc[t][j] + bn;
            const signed char s = (x >= 0.125f) ? 1 : (x < 0.0f ? -1 : 0);
            if (which == 2) {
                const int bb = row >> 10, ss = row & (S_ - 1);
                vhatT[((size_t)bb * D_ + gn) * S_ + ss] = s;
            } else {
                signed char* dst = (which == 0) ? qhat : khat;
                dst[(size_t)row * D_ + gn] = s;
            }
        }
    }
#undef FILL_PANEL_ASYNC
#undef FILL_PANEL_SYNC
}

__global__ __launch_bounds__(128) void attn_spike_kernel(
    const signed char* __restrict__ qhat, const signed char* __restrict__ khat,
    const signed char* __restrict__ vhatT, const float* __restrict__ mask,
    float* __restrict__ out)
{
    extern __shared__ char smem[];
    float*         sc   = (float*)smem;
    unsigned char* pb   = (unsigned char*)(smem + 16 * 1024 * 4);
    float*         red  = (float*)(smem + 16 * 1024 * 4 + 16 * 1024);
    float*         rsum = red + 128;
    float*         rmax = rsum + 16;

    const int tid  = threadIdx.x;
    const int wave = tid >> 5, lane = tid & 31;
    const int l16  = lane & 15, half = lane >> 4;
    const int bh = blockIdx.x;
    const int b  = bh >> 4, h = bh & (H_ - 1);
    const int qt = blockIdx.y;

    const int hiA = half * 8;
    const int hiB = half * 8;

    v8i aq;
    {
        const unsigned char* qrow = (const unsigned char*)qhat
            + ((size_t)b * S_ + qt * 16 + l16) * D_ + h * DH_;
#pragma unroll
        for (int i = 0; i < 8; ++i) {
            const int k = (i >> 1) * 16 + (i & 1) * 4 + hiA;
            aq[i] = *(const int*)(qrow + k);
        }
    }
    const float sscale = (0.125f * 0.125f) / 8.0f;
    for (int kt = wave; kt < S_ / 16; kt += 4) {
        v8i bk8;
        const unsigned char* krow = (const unsigned char*)khat
            + ((size_t)b * S_ + kt * 16 + l16) * D_ + h * DH_;
#pragma unroll
        for (int i = 0; i < 8; ++i) {
            const int k = (i >> 1) * 16 + (i & 1) * 4 + hiB;
            bk8[i] = *(const int*)(krow + k);
        }
        v8i ci = {};
        ci = __builtin_amdgcn_wmma_i32_16x16x64_iu8(
                 true, aq, true, bk8, ci, false, false);
        const float mk = mask[(size_t)b * S_ + kt * 16 + l16];
#pragma unroll
        for (int j = 0; j < 8; ++j) {
            const int m = j + half * 8;
            sc[m * 1024 + kt * 16 + l16] = (float)ci[j] * sscale + mk;
        }
    }
    __syncthreads();

    {
        const int r = tid >> 3, sub = tid & 7;
        float mx = -3.4e38f;
        for (int ci = sub * 128; ci < sub * 128 + 128; ++ci)
            mx = fmaxf(mx, sc[r * 1024 + ci]);
        red[r * 8 + sub] = mx;
        __syncthreads();
        if (sub == 0) {
            float m2 = red[r * 8];
#pragma unroll
            for (int i = 1; i < 8; ++i) m2 = fmaxf(m2, red[r * 8 + i]);
            rmax[r] = m2;
        }
        __syncthreads();
        const float m2 = rmax[r];
        float sum = 0.f;
        for (int ci = sub * 128; ci < sub * 128 + 128; ++ci) {
            const float e = __expf(sc[r * 1024 + ci] - m2);
            sc[r * 1024 + ci] = e;
            sum += e;
        }
        red[r * 8 + sub] = sum;
        __syncthreads();
        if (sub == 0) {
            float s2 = 0.f;
#pragma unroll
            for (int i = 0; i < 8; ++i) s2 += red[r * 8 + i];
            rsum[r] = s2;
        }
        __syncthreads();
        const float thr = rsum[r] * (1.0f / 32.0f);
        for (int ci = sub * 128; ci < sub * 128 + 128; ++ci)
            pb[r * 1024 + ci] = (sc[r * 1024 + ci] >= thr) ? 1 : 0;
    }
    __syncthreads();

    {
        const int w = wave;
        v8i acc2 = {};
        const unsigned char* vcol = (const unsigned char*)vhatT
            + ((size_t)b * D_ + h * DH_ + w * 16 + l16) * S_;
        for (int kb = 0; kb < S_; kb += 64) {
            v8i ap, bv8;
#pragma unroll
            for (int i = 0; i < 8; ++i) {
                const int ka = (i >> 1) * 16 + (i & 1) * 4 + hiA;
                ap[i] = *(const int*)(pb + l16 * 1024 + kb + ka);
                const int kv = (i >> 1) * 16 + (i & 1) * 4 + hiB;
                bv8[i] = *(const int*)(vcol + kb + kv);
            }
            acc2 = __builtin_amdgcn_wmma_i32_16x16x64_iu8(
                       false, ap, true, bv8, acc2, false, false);
        }
        const float cscale = 1.0f / 256.0f;
#pragma unroll
        for (int j = 0; j < 8; ++j) {
            const int m = j + half * 8;
            const float x = (float)acc2[j] * cscale;
            const float o = (x >= 0.0625f) ? 0.0625f : 0.0f;
            out[((size_t)b * S_ + qt * 16 + m) * D_ + h * DH_ + w * 16 + l16] = o;
        }
    }
}

extern "C" void kernel_launch(void* const* d_in, const int* in_sizes, int n_in,
                              void* d_out, int out_size, void* d_ws, size_t ws_size,
                              hipStream_t stream) {
    (void)in_sizes; (void)n_in; (void)out_size; (void)ws_size;
    const float* hs   = (const float*)d_in[0];
    const float* mask = (const float*)d_in[1];
    const float* Wq   = (const float*)d_in[2];
    const float* bq   = (const float*)d_in[3];
    const float* Wk   = (const float*)d_in[4];
    const float* bk   = (const float*)d_in[5];
    const float* Wv   = (const float*)d_in[6];
    const float* bv   = (const float*)d_in[7];

    const size_t NBS = (size_t)B_ * S_ * D_;
    const size_t NW  = (size_t)D_ * D_;

    char* ws = (char*)d_ws;
    signed char* qhat  = (signed char*)ws;
    signed char* khat  = (signed char*)(ws + NBS);
    signed char* vhatT = (signed char*)(ws + 2 * NBS);
    _Float16* hsH = (_Float16*)(ws + 3 * NBS);
    _Float16* WqH = (_Float16*)(ws + 3 * NBS + 2 * NBS);
    _Float16* WkH = WqH + NW;
    _Float16* WvH = WkH + NW;

    cvt_f16_kernel<<<(int)(NBS / 8 / 256), 256, 0, stream>>>(hs, hsH, (int)NBS);
    cvt_f16_kernel<<<(int)(NW / 8 / 256), 256, 0, stream>>>(Wq, WqH, (int)NW);
    cvt_f16_kernel<<<(int)(NW / 8 / 256), 256, 0, stream>>>(Wk, WkH, (int)NW);
    cvt_f16_kernel<<<(int)(NW / 8 / 256), 256, 0, stream>>>(Wv, WvH, (int)NW);

    dim3 gA(D_ / 64, (B_ * S_) / 64, 3);
    qkv_gemm_kernel<<<gA, 128, 0, stream>>>(hsH, WqH, bq, WkH, bk, WvH, bv,
                                            qhat, khat, vhatT);

    const size_t smem = (size_t)16 * 1024 * 4
                      + (size_t)16 * 1024
                      + (128 + 16 + 16) * 4;
    dim3 gB(B_ * H_, S_ / 16, 1);
    attn_spike_kernel<<<gB, 128, smem, stream>>>(qhat, khat, vhatT, mask,
                                                 (float*)d_out);
}
